// Net_58729382805606
// MI455X (gfx1250) — matrix operand order repaired and checked by reading, not compiled, not run on hardware
//
#include <hip/hip_runtime.h>
#include <hip/hip_bf16.h>
#include <math.h>

// ---------------------------------------------------------------------------
// 2-layer GCN for MI455X (gfx1250, wave32).
//   layer GEMMs  -> v_wmma_f32_16x16x32_bf16 (bf16 inputs, f32 accumulate)
//   aggregation  -> global_atomic_add_f32 (h fits in 192MB L2, so L2 traffic)
//   softmax/orth -> scalar VALU kernels
// ---------------------------------------------------------------------------

typedef __bf16 bf16_t;
typedef bf16_t v16bf __attribute__((ext_vector_type(16)));
typedef float  v8f   __attribute__((ext_vector_type(8)));

// ---------------- WMMA GEMM: D[nrows x NOUT] = A[nrows x K] * B[K x NOUT] ---
// A, B, D fp32 row-major. One wave computes a 16-row slab x NTILES*16 cols.
// Block = 128 threads = 4 waves = 64 rows.
template<int K, int NOUT, int NTILES>
__global__ __launch_bounds__(128)
void gemm_wmma_bf16(const float* __restrict__ A, const float* __restrict__ B,
                    float* __restrict__ D, int nrows)
{
    const int lane = threadIdx.x & 31;
    const int wave = threadIdx.x >> 5;
    const int row0 = blockIdx.x * 64 + wave * 16;
    if (row0 >= nrows) return;                 // wave-uniform (nrows % 16 == 0)

    const int m  = lane & 15;                  // A: row within tile / B,D: col
    const int hi = lane >> 4;                  // half-wave select

    const float* __restrict__ arow = A + (size_t)(row0 + m) * K;

    v8f acc[NTILES];
    #pragma unroll
    for (int t = 0; t < NTILES; ++t)
        #pragma unroll
        for (int r = 0; r < 8; ++r) acc[t][r] = 0.0f;

    for (int kt = 0; kt < K / 32; ++kt) {
        // A fragment: lane holds row m, K = kt*32 + hi*8 + {0..7, 16..23}
        v16bf afrag;
        const float* ap = arow + kt * 32 + hi * 8;
        #pragma unroll
        for (int e = 0; e < 8; ++e) {
            afrag[e]     = (bf16_t)ap[e];       // K offset hi*8 + e
            afrag[e + 8] = (bf16_t)ap[e + 16];  // K offset 16 + hi*8 + e
        }
        #pragma unroll
        for (int t = 0; t < NTILES; ++t) {
            const int col = t * 16 + m;
            // B fragment: lane holds column col, K = kt*32 + hi*8 + {0..7, 16..23}
            v16bf bfrag;
            #pragma unroll
            for (int e = 0; e < 16; ++e) {
                const int k = kt * 32 + hi * 8 + e + (e >= 8 ? 8 : 0);
                float v = 0.0f;
                if (col < NOUT) v = B[(size_t)k * NOUT + col];
                bfrag[e] = (bf16_t)v;
            }
            acc[t] = __builtin_amdgcn_wmma_f32_16x16x32_bf16(
                false, afrag, false, bfrag, (short)0, acc[t], false, false);
        }
    }

    // D store: VGPR r of accumulator -> row (hi*8 + r), column col
    #pragma unroll
    for (int t = 0; t < NTILES; ++t) {
        const int col = t * 16 + m;
        if (col < NOUT) {
            #pragma unroll
            for (int r = 0; r < 8; ++r)
                D[(size_t)(row0 + hi * 8 + r) * NOUT + col] = acc[t][r];
        }
    }
}

// ---------------- degree / normalization ----------------------------------
__global__ void k_fill1(float* __restrict__ a, int n)
{
    int i = blockIdx.x * blockDim.x + threadIdx.x;
    if (i < n) a[i] = 1.0f;                    // self-loop contribution
}

__global__ void k_deg_accum(const long long* __restrict__ dst,
                            float* __restrict__ deg, int e)
{
    int i = blockIdx.x * blockDim.x + threadIdx.x;
    if (i < e) atomicAdd(&deg[(int)dst[i]], 1.0f);
}

__global__ void k_rsqrt_inplace(float* __restrict__ a, int n)
{
    int i = blockIdx.x * blockDim.x + threadIdx.x;
    if (i < n) a[i] = rsqrtf(fmaxf(a[i], 1.0f));
}

// ---------------- edge aggregation -----------------------------------------
// self-loop term: agg[i] = t[i] * dinv[i]^2
template<int F>
__global__ void k_self_init(const float* __restrict__ t,
                            const float* __restrict__ dinv,
                            float* __restrict__ agg, int n)
{
    long long idx = (long long)blockIdx.x * blockDim.x + threadIdx.x;
    if (idx >= (long long)n * F) return;
    int i = (int)(idx / F);
    float w = dinv[i];
    agg[idx] = t[idx] * w * w;
}

// edge term: agg[dst] += t[src] * dinv[src]*dinv[dst]   (one thread per feat)
template<int F>
__global__ void k_edge_agg(const long long* __restrict__ src,
                           const long long* __restrict__ dst,
                           const float* __restrict__ dinv,
                           const float* __restrict__ t,
                           float* __restrict__ agg, int e)
{
    long long idx = (long long)blockIdx.x * blockDim.x + threadIdx.x;
    int ei = (int)(idx / F);
    int f  = (int)(idx % F);
    if (ei >= e) return;
    int s = (int)src[ei];
    int d = (int)dst[ei];
    float w = dinv[s] * dinv[d];
    atomicAdd(&agg[(size_t)d * F + f], t[(size_t)s * F + f] * w);
}

template<int F>
__global__ void k_bias_relu(float* __restrict__ hb, const float* __restrict__ b,
                            int n)
{
    long long idx = (long long)blockIdx.x * blockDim.x + threadIdx.x;
    if (idx >= (long long)n * F) return;
    int c = (int)(idx % F);
    hb[idx] = fmaxf(hb[idx] + b[c], 0.0f);
}

// ---------------- output: bias + log_softmax + raw logits ------------------
template<int C>
__global__ void k_out_logsoftmax(const float* __restrict__ agg,
                                 const float* __restrict__ b,
                                 float* __restrict__ out, int n)
{
    int i = blockIdx.x * blockDim.x + threadIdx.x;
    if (i >= n) return;
    float v[C];
    float mx = -__builtin_inff();
    #pragma unroll
    for (int c = 0; c < C; ++c) {
        v[c] = agg[(size_t)i * C + c] + b[c];
        mx = fmaxf(mx, v[c]);
    }
    float s = 0.0f;
    #pragma unroll
    for (int c = 0; c < C; ++c) s += expf(v[c] - mx);
    float lse = logf(s) + mx;
    float* __restrict__ xo = out + (size_t)n * C;   // second output: x_out
    #pragma unroll
    for (int c = 0; c < C; ++c) {
        out[(size_t)i * C + c] = v[c] - lse;        // first output: log_softmax
        xo[(size_t)i * C + c]  = v[c];
    }
}

// ---------------- orthogonality penalty ------------------------------------
__global__ void k_zero_sums(float* __restrict__ s) { s[0] = 0.0f; s[1] = 0.0f; }

// sum over (i,j) of (W[i,:].W[j,:] - delta_ij)^2 ; wave-reduced atomics
template<int DIM, int INNER>
__global__ void k_orth(const float* __restrict__ W, float* __restrict__ sum)
{
    int idx = blockIdx.x * blockDim.x + threadIdx.x;
    float d2 = 0.0f;
    if (idx < DIM * DIM) {
        int i = idx / DIM, j = idx % DIM;
        const float* wi = W + (size_t)i * INNER;
        const float* wj = W + (size_t)j * INNER;
        float acc = 0.0f;
        #pragma unroll 8
        for (int k = 0; k < INNER; ++k) acc = fmaf(wi[k], wj[k], acc);
        float d = acc - (i == j ? 1.0f : 0.0f);
        d2 = d * d;
    }
    #pragma unroll
    for (int off = 16; off > 0; off >>= 1) d2 += __shfl_down(d2, off, 32);
    if ((threadIdx.x & 31) == 0) atomicAdd(sum, d2);
}

__global__ void k_orth_final(const float* __restrict__ s, float* __restrict__ out)
{
    out[0] = sqrtf(s[0]) + sqrtf(s[1]);
}

// ---------------------------------------------------------------------------
extern "C" void kernel_launch(void* const* d_in, const int* in_sizes, int n_in,
                              void* d_out, int out_size, void* d_ws, size_t ws_size,
                              hipStream_t stream)
{
    (void)n_in; (void)out_size; (void)ws_size;

    const float*     x  = (const float*)d_in[0];       // [N, 512]
    const long long* ei = (const long long*)d_in[1];   // [2, E] int64
    const float*     W1 = (const float*)d_in[2];       // [512, 64]
    const float*     b1 = (const float*)d_in[3];       // [64]
    const float*     W2 = (const float*)d_in[4];       // [64, 40]
    const float*     b2 = (const float*)d_in[5];       // [40]

    const int FIN = 512, HID = 64, C = 40;
    const int N = in_sizes[0] / FIN;
    const int E = in_sizes[1] / 2;
    const long long* src = ei;
    const long long* dst = ei + E;

    // workspace layout (floats)
    float* ws   = (float*)d_ws;
    float* dinv = ws;                           // N        (deg -> dinv in place)
    float* t1   = dinv + N;                     // N*HID    x @ W1
    float* h    = t1 + (size_t)N * HID;         // N*HID    agg1 -> relu(h)
    float* t2   = h  + (size_t)N * HID;         // N*C      h @ W2
    float* agg2 = t2 + (size_t)N * C;           // N*C
    float* sums = agg2 + (size_t)N * C;         // 2

    float* out = (float*)d_out;                 // [N*C log_softmax | N*C x_out | 1 orth]

    auto cdiv = [](long long a, long long b) { return (unsigned)((a + b - 1) / b); };
    dim3 blk(256);

    k_zero_sums<<<1, 1, 0, stream>>>(sums);

    // degree + symmetric normalization (self-loops included)
    k_fill1        <<<cdiv(N, 256), blk, 0, stream>>>(dinv, N);
    k_deg_accum    <<<cdiv(E, 256), blk, 0, stream>>>(dst, dinv, E);
    k_rsqrt_inplace<<<cdiv(N, 256), blk, 0, stream>>>(dinv, N);

    // ---- layer 1: t1 = x @ W1 ; h = relu(Ahat t1 + b1) ----
    gemm_wmma_bf16<512, 64, 4><<<cdiv(N, 64), 128, 0, stream>>>(x, W1, t1, N);
    k_self_init<64><<<cdiv((long long)N * 64, 256), blk, 0, stream>>>(t1, dinv, h, N);
    k_edge_agg<64> <<<cdiv((long long)E * 64, 256), blk, 0, stream>>>(src, dst, dinv, t1, h, E);
    k_bias_relu<64><<<cdiv((long long)N * 64, 256), blk, 0, stream>>>(h, b1, N);

    // ---- layer 2: t2 = h @ W2 ; x_out = Ahat t2 + b2 ----
    gemm_wmma_bf16<64, 40, 3><<<cdiv(N, 64), 128, 0, stream>>>(h, W2, t2, N);
    k_self_init<40><<<cdiv((long long)N * 40, 256), blk, 0, stream>>>(t2, dinv, agg2, N);
    k_edge_agg<40> <<<cdiv((long long)E * 40, 256), blk, 0, stream>>>(src, dst, dinv, t2, agg2, E);
    k_out_logsoftmax<40><<<cdiv(N, 256), blk, 0, stream>>>(agg2, b2, out, N);

    // ---- orthogonality penalty ----
    k_orth<512, 64><<<cdiv(512 * 512, 256), blk, 0, stream>>>(W1, &sums[0]);
    k_orth<64, 40> <<<cdiv(64 * 64, 256),   blk, 0, stream>>>(W2, &sums[1]);
    k_orth_final<<<1, 1, 0, stream>>>(sums, out + (size_t)2 * N * C);
}
